// AttentionBlock_38706245271924
// MI455X (gfx1250) — matrix operand order repaired and checked by reading, not compiled, not run on hardware
//
#include <hip/hip_runtime.h>
#include <hip/hip_bf16.h>

typedef __attribute__((ext_vector_type(16))) _Float16 v16h;
typedef __attribute__((ext_vector_type(8)))  float    v8f;
typedef unsigned int v4u __attribute__((ext_vector_type(4)));
typedef int          v8i_t __attribute__((ext_vector_type(8)));
typedef int          v4i_t __attribute__((ext_vector_type(4)));

#define WMMA_F16(A, B, Cc) \
  __builtin_amdgcn_wmma_f32_16x16x32_f16(false, (A), false, (B), (short)0, (Cc), false, false)

struct U32x8 { uint4 lo, hi; };

// A-layout loader: elements 0..7 from p0 (16B), elements 8..15 from p1 (16B)
__device__ inline v16h ld_a(const _Float16* p0, const _Float16* p1) {
  U32x8 t;
  t.lo = *(const uint4*)p0;
  t.hi = *(const uint4*)p1;
  return __builtin_bit_cast(v16h, t);
}

// B-layout loader: 16 contiguous halves (32B) = one Bᵀ-row chunk
__device__ inline v16h ld_b(const _Float16* p) {
  U32x8 t;
  t.lo = *(const uint4*)p;
  t.hi = *(const uint4*)(p + 8);
  return __builtin_bit_cast(v16h, t);
}

#define NB 4
#define HW 4096   // 64*64 pixels per batch
#define CH 64
#define NGRP 32

#if defined(__gfx1250__) && __has_builtin(__builtin_amdgcn_tensor_load_to_lds)
#define USE_TDM 1
#else
#define USE_TDM 0
#endif

#if USE_TDM
// Issue a 2-D TDM tile load (data_size = 2 bytes). Descriptor packing per
// CDNA5 ISA section 8 (D# groups 0/1; groups 2/3 zero => 2-D tensor).
// This toolchain's builtin takes 6 args: (g0, g1, g2, g3, g_extra, cpol).
__device__ inline void tdm_load_2d(unsigned lds_addr, const void* gaddr,
                                   unsigned tensor_d0, unsigned tensor_d1,
                                   unsigned tile_d0, unsigned tile_d1,
                                   unsigned long long stride0_elems) {
  unsigned long long ga = (unsigned long long)(size_t)gaddr;
  v4u g0;
  g0[0] = 1u;                                            // count=1, user mode
  g0[1] = lds_addr;                                      // lds_addr[31:0]
  g0[2] = (unsigned)(ga & 0xFFFFFFFFu);                  // global_addr[31:0]
  g0[3] = (unsigned)((ga >> 32) & 0x1FFFFFFu) | (2u << 30); // addr[56:32] | type=2

  unsigned long long q0 = (1ull << 16)                            // data_size=1 (2B)
                        | ((unsigned long long)(tensor_d0 & 0xFFFFu) << 48);
  unsigned long long q1 = (unsigned long long)(tensor_d0 >> 16)
                        | ((unsigned long long)tensor_d1 << 16)   // tensor_dim1 [111:80]
                        | ((unsigned long long)(tile_d0 & 0xFFFFu) << 48); // tile_dim0
  unsigned long long q2 = (unsigned long long)(tile_d1 & 0xFFFFu)          // tile_dim1
                        | ((stride0_elems & 0xFFFFFFFFull) << 32);         // stride0[31:0]
  unsigned long long q3 = (stride0_elems >> 32) & 0xFFFFull;               // stride0[47:32]

  v8i_t g1;
  g1[0] = (int)(unsigned)q0;  g1[1] = (int)(unsigned)(q0 >> 32);
  g1[2] = (int)(unsigned)q1;  g1[3] = (int)(unsigned)(q1 >> 32);
  g1[4] = (int)(unsigned)q2;  g1[5] = (int)(unsigned)(q2 >> 32);
  g1[6] = (int)(unsigned)q3;  g1[7] = (int)(unsigned)(q3 >> 32);
  v4i_t z4 = {0, 0, 0, 0};
  v8i_t z8 = {0, 0, 0, 0, 0, 0, 0, 0};
  __builtin_amdgcn_tensor_load_to_lds(g0, g1, z4, z4, z8, 0);
}
#endif

// Projection helper: D = A x Wᵀ-tile + bias, row-major f16 destination.
// dst points to the tile's pixel-0 row. Stores use immediate offsets (e*CH).
__device__ inline void proj_rm(const v16h* A, const _Float16* wt, const float* bias,
                               _Float16* dst, int ln, int half, float scl) {
  for (int t = 0; t < 4; ++t) {
    float bb = bias[t * 16 + ln];
    v8f acc;
    for (int e = 0; e < 8; ++e) acc[e] = bb;
    for (int ks = 0; ks < 2; ++ks)
      acc = WMMA_F16(A[ks], ld_b(wt + (size_t)(t * 16 + ln) * CH + ks * 32 + half * 16), acc);
    _Float16* d = dst + (size_t)(half * 8) * CH + t * 16 + ln;
    for (int e = 0; e < 8; ++e)
      d[(size_t)e * CH] = (_Float16)(acc[e] * scl);
  }
}

// Projection helper with transposed destination ([channel][pixel], pixel
// contiguous): one packed 16B store per n-tile.
__device__ inline void proj_tr(const v16h* A, const _Float16* wt, const float* bias,
                               _Float16* dst, int ln, int half) {
  for (int t = 0; t < 4; ++t) {
    float bb = bias[t * 16 + ln];
    v8f acc;
    for (int e = 0; e < 8; ++e) acc[e] = bb;
    for (int ks = 0; ks < 2; ++ks)
      acc = WMMA_F16(A[ks], ld_b(wt + (size_t)(t * 16 + ln) * CH + ks * 32 + half * 16), acc);
    _Float16 tmp[8];
    for (int e = 0; e < 8; ++e) tmp[e] = (_Float16)acc[e];
    _Float16* d = dst + (size_t)(t * 16 + ln) * HW + half * 8;
    *(uint4*)d = *(const uint4*)tmp;
  }
}

// ---------------------------------------------------------------------------
// Kernel 0: pack weights into f16 transposed form. Wt[m][n][k] = w_m[k][n].
// ---------------------------------------------------------------------------
__global__ __launch_bounds__(256) void prep_weights_kernel(
    const float* __restrict__ wq, const float* __restrict__ wk,
    const float* __restrict__ wv, const float* __restrict__ wo,
    _Float16* __restrict__ Wt) {
  const float* src[4] = {wq, wk, wv, wo};
  const float* w = src[blockIdx.x];
  _Float16* dst = Wt + (size_t)blockIdx.x * CH * CH;
  for (int j = 0; j < 16; ++j) {
    int idx = threadIdx.x * 16 + j;        // 0..4095
    int n = idx >> 6, k = idx & 63;
    dst[n * CH + k] = (_Float16)w[k * CH + n];
  }
}

// ---------------------------------------------------------------------------
// Kernel 1: GroupNorm statistics. One block per (batch, group): 8192 elements.
// ---------------------------------------------------------------------------
__global__ __launch_bounds__(256) void gn_stats_kernel(const float* __restrict__ x,
                                                       float* __restrict__ stats) {
  const int bg = blockIdx.x;            // 0..127
  const int b = bg >> 5, g = bg & 31;
  const float* xb = x + (size_t)b * HW * CH + g * 2;
  float s = 0.f, ss = 0.f;
  for (int p = threadIdx.x; p < HW; p += 256) {
    float a0 = xb[(size_t)p * CH];
    float a1 = xb[(size_t)p * CH + 1];
    s  += a0 + a1;
    ss += a0 * a0 + a1 * a1;
  }
  __shared__ float sh[512];
  sh[threadIdx.x] = s;
  sh[256 + threadIdx.x] = ss;
  __syncthreads();
  for (int o = 128; o > 0; o >>= 1) {
    if ((int)threadIdx.x < o) {
      sh[threadIdx.x]       += sh[threadIdx.x + o];
      sh[256 + threadIdx.x] += sh[256 + threadIdx.x + o];
    }
    __syncthreads();
  }
  if (threadIdx.x == 0) {
    float mean = sh[0] * (1.0f / 8192.0f);
    float var  = sh[256] * (1.0f / 8192.0f) - mean * mean;
    stats[bg * 2]     = mean;
    stats[bg * 2 + 1] = rsqrtf(var + 1e-3f);
  }
}

// ---------------------------------------------------------------------------
// Kernel 2: GroupNorm-apply + Q/K/V projections via WMMA.
// Q is pre-scaled by C^-0.5 = 0.125 so the attention loop skips score scaling.
// ---------------------------------------------------------------------------
__global__ __launch_bounds__(128) void qkv_kernel(
    const float* __restrict__ x, const float* __restrict__ gamma, const float* __restrict__ beta,
    const float* __restrict__ bq, const float* __restrict__ bk, const float* __restrict__ bv,
    const float* __restrict__ stats, const _Float16* __restrict__ Wt,
    _Float16* __restrict__ Q, _Float16* __restrict__ K, _Float16* __restrict__ Vt) {
  const int wave = threadIdx.x >> 5, lane = threadIdx.x & 31;
  const int tile = blockIdx.x * 4 + wave;      // 0..1023
  const int b  = tile >> 8;
  const int tb = (tile & 255) << 4;
  const int ln = lane & 15, half = lane >> 4;

  // Normalized activations in A layout (16 rows x 64 ch, two K=32 steps)
  const float* xrow = x + ((size_t)b * HW + tb + ln) * CH;
  const float* st = stats + b * NGRP * 2;
  v16h A[2];
  for (int ks = 0; ks < 2; ++ks) {
    for (int chunk = 0; chunk < 2; ++chunk) {
      int cb = ks * 32 + half * 16 + chunk * 8;
      for (int j = 0; j < 8; ++j) {
        int c = cb + j;
        float mean = st[(c >> 1) * 2];
        float rstd = st[(c >> 1) * 2 + 1];
        float v = (xrow[c] - mean) * rstd * gamma[c] + beta[c];
        A[ks][chunk * 8 + j] = (_Float16)v;
      }
    }
  }

  proj_rm(A, Wt,                bq, Q + ((size_t)b * HW + tb) * CH, ln, half, 0.125f);
  proj_rm(A, Wt + 1 * CH * CH,  bk, K + ((size_t)b * HW + tb) * CH, ln, half, 1.0f);
  proj_tr(A, Wt + 2 * CH * CH,  bv, Vt + (size_t)b * CH * HW + tb,  ln, half);
}

// ---------------------------------------------------------------------------
// Kernel 3: flash attention (online softmax) + output projection + residual.
// Block = 4 waves = 4 query tiles (same batch); K/V tiles double-buffered in
// LDS via the Tensor Data Mover (fallback: cooperative copy).
// ---------------------------------------------------------------------------
__global__ __launch_bounds__(128) void attn_kernel(
    const float* __restrict__ x,
    const _Float16* __restrict__ Q, const _Float16* __restrict__ Km,
    const _Float16* __restrict__ Vt, const _Float16* __restrict__ Wt,
    const float* __restrict__ bo, float* __restrict__ out) {
  __shared__ _Float16 Kl[2][32][64];    // 2 x (32 keys x 64 ch)   (8KB)
  __shared__ _Float16 Vl[2][64][32];    // 2 x (64 ch x 32 keys)   (8KB)
  __shared__ _Float16 Pb[4][16][64];    // per-wave transpose      (8KB)

  const int wave = threadIdx.x >> 5, lane = threadIdx.x & 31;
  const int tile = blockIdx.x * 4 + wave;
  const int b  = tile >> 8;
  const int tb = (tile & 255) << 4;
  const int ln = lane & 15, half = lane >> 4;
  _Float16(*P)[64] = Pb[wave];

  const _Float16* Kb_ = Km + (size_t)b * HW * CH;
  const _Float16* Vb_ = Vt + (size_t)b * CH * HW;

  // Q A-tiles, reused across all 128 key steps
  const _Float16* qrow = Q + ((size_t)b * HW + tb + ln) * CH;
  v16h QA[2];
  for (int ks = 0; ks < 2; ++ks)
    QA[ks] = ld_a(qrow + ks * 32 + half * 16, qrow + ks * 32 + half * 16 + 8);

  v16h ones;
  for (int j = 0; j < 16; ++j) ones[j] = (_Float16)1.0f;

  v8f Oc[4], Ol;
  for (int t = 0; t < 4; ++t)
    for (int e = 0; e < 8; ++e) Oc[t][e] = 0.f;
  for (int e = 0; e < 8; ++e) Ol[e] = 0.f;
  float m[8];
  for (int r = 0; r < 8; ++r) m[r] = -1e30f;

#if USE_TDM
  if (wave == 0) {    // prologue: stage tile 0 into buffer 0
    tdm_load_2d((unsigned)(size_t)&Kl[0][0][0], Kb_,
                CH, HW, CH, 32, CH);
    tdm_load_2d((unsigned)(size_t)&Vl[0][0][0], Vb_,
                HW, CH, 32, CH, HW);
  }
#endif

  for (int i = 0; i < HW / 32; ++i) {
    const int kb = i * 32;
    const int cur = i & 1;
    __syncthreads();                       // buffers from iter i-1 fully read
#if USE_TDM
    if (wave == 0) {
      if (i + 1 < HW / 32) {
        const int nxt = cur ^ 1;
        tdm_load_2d((unsigned)(size_t)&Kl[nxt][0][0], Kb_ + (size_t)(kb + 32) * CH,
                    CH, HW, CH, 32, CH);
        tdm_load_2d((unsigned)(size_t)&Vl[nxt][0][0], Vb_ + kb + 32,
                    HW, CH, 32, CH, HW);
        __builtin_amdgcn_s_wait_tensorcnt(2);   // tile i done, tile i+1 in flight
      } else {
        __builtin_amdgcn_s_wait_tensorcnt(0);   // last tile done
      }
    }
#else
    {
      // cooperative copy of tile i into buffer cur: 128 threads x 32B chunks
      int ti = threadIdx.x;
      const uint4* gk = (const uint4*)(Kb_ + (size_t)(kb + (ti >> 1)) * CH + (ti & 1) * 32);
      uint4* lk = (uint4*)(&Kl[cur][ti >> 1][(ti & 1) * 32]);
      lk[0] = gk[0]; lk[1] = gk[1];
      const uint4* gv = (const uint4*)(Vb_ + (size_t)(ti >> 1) * HW + kb + (ti & 1) * 16);
      uint4* lv = (uint4*)(&Vl[cur][ti >> 1][(ti & 1) * 16]);
      lv[0] = gv[0]; lv[1] = gv[1];
    }
#endif
    __syncthreads();

    // S = Q Kᵀ (16q x 32k) in two C-layout accumulators (scale folded into Q)
    v8f S0, S1;
    for (int e = 0; e < 8; ++e) { S0[e] = 0.f; S1[e] = 0.f; }
    for (int ks = 0; ks < 2; ++ks) {
      S0 = WMMA_F16(QA[ks], ld_b(&Kl[cur][ln][ks * 32 + half * 16]), S0);
      S1 = WMMA_F16(QA[ks], ld_b(&Kl[cur][16 + ln][ks * 32 + half * 16]), S1);
    }

    // row-max over this lane's pair, reduce across half-wave
    float rmax[8];
    for (int r = 0; r < 8; ++r) rmax[r] = fmaxf(S0[r], S1[r]);
    for (int off = 1; off < 16; off <<= 1)
      for (int r = 0; r < 8; ++r)
        rmax[r] = fmaxf(rmax[r], __shfl_xor(rmax[r], off, 32));

    for (int r = 0; r < 8; ++r) {
      float mnew = fmaxf(m[r], rmax[r]);
      float corr = __expf(m[r] - mnew);
      m[r] = mnew;
      float p0 = __expf(S0[r] - mnew);
      float p1 = __expf(S1[r] - mnew);
      P[r + half * 8][ln]      = (_Float16)p0;
      P[r + half * 8][16 + ln] = (_Float16)p1;
      Ol[r] *= corr;                           // l rides along as ones-column WMMA
      for (int t = 0; t < 4; ++t) Oc[t][r] *= corr;
    }
    asm volatile("s_wait_dscnt 0x0" ::: "memory");   // same-wave LDS store->load

    v16h PA = ld_a(&P[ln][half * 16], &P[ln][half * 16 + 8]);
    for (int t = 0; t < 4; ++t)
      Oc[t] = WMMA_F16(PA, ld_b(&Vl[cur][t * 16 + ln][half * 16]), Oc[t]);
    Ol = WMMA_F16(PA, ones, Ol);               // row-sum of P via WMMA
  }

  // Normalize, transpose via LDS, project with wo, add bias + residual
  for (int r = 0; r < 8; ++r) {
    float inv = 1.0f / Ol[r];
    for (int t = 0; t < 4; ++t)
      P[r + half * 8][t * 16 + ln] = (_Float16)(Oc[t][r] * inv);
  }
  asm volatile("s_wait_dscnt 0x0" ::: "memory");

  v16h OA[2];
  for (int ks = 0; ks < 2; ++ks)
    OA[ks] = ld_a(&P[ln][ks * 32 + half * 16], &P[ln][ks * 32 + half * 16 + 8]);

  const _Float16* wt_o = Wt + (size_t)3 * CH * CH;
  for (int t = 0; t < 4; ++t) {
    float bb = bo[t * 16 + ln];
    v8f acc;
    for (int e = 0; e < 8; ++e) acc[e] = bb;
    for (int ks = 0; ks < 2; ++ks)
      acc = WMMA_F16(OA[ks], ld_b(wt_o + (size_t)(t * 16 + ln) * CH + ks * 32 + half * 16), acc);
    const float* xb2 = x   + ((size_t)b * HW + tb + half * 8) * CH + t * 16 + ln;
    float*       ob  = out + ((size_t)b * HW + tb + half * 8) * CH + t * 16 + ln;
    for (int e = 0; e < 8; ++e)
      ob[(size_t)e * CH] = xb2[(size_t)e * CH] + acc[e];
  }
}

// ---------------------------------------------------------------------------
extern "C" void kernel_launch(void* const* d_in, const int* in_sizes, int n_in,
                              void* d_out, int out_size, void* d_ws, size_t ws_size,
                              hipStream_t stream) {
  const float* x     = (const float*)d_in[0];
  const float* gamma = (const float*)d_in[1];
  const float* beta  = (const float*)d_in[2];
  const float* wq    = (const float*)d_in[3];
  const float* bq    = (const float*)d_in[4];
  const float* wk    = (const float*)d_in[5];
  const float* bk    = (const float*)d_in[6];
  const float* wv    = (const float*)d_in[7];
  const float* bv    = (const float*)d_in[8];
  const float* wo    = (const float*)d_in[9];
  const float* bo    = (const float*)d_in[10];
  float* out = (float*)d_out;

  const size_t mat_elems = (size_t)NB * HW * CH;   // 1,048,576 halves each
  _Float16* Q  = (_Float16*)d_ws;
  _Float16* K  = Q + mat_elems;
  _Float16* Vt = K + mat_elems;
  _Float16* Wt = Vt + mat_elems;                   // 4 x 64 x 64 f16
  float* stats = (float*)(Wt + 4 * CH * CH);       // 256 floats

  prep_weights_kernel<<<4, 256, 0, stream>>>(wq, wk, wv, wo, Wt);
  gn_stats_kernel<<<NB * NGRP, 256, 0, stream>>>(x, stats);
  qkv_kernel<<<256, 128, 0, stream>>>(x, gamma, beta, bq, bk, bv, stats, Wt, Q, K, Vt);
  attn_kernel<<<256, 128, 0, stream>>>(x, Q, K, Vt, Wt, bo, out);
}
